// OnLSTM_79267916415624
// MI455X (gfx1250) — matrix operand order repaired and checked by reading, not compiled, not run on hardware
//
#include <hip/hip_runtime.h>
#include <hip/hip_bf16.h>

typedef __attribute__((ext_vector_type(16))) __bf16 v16bf;
typedef __attribute__((ext_vector_type(8)))  __bf16 v8bf;
typedef __attribute__((ext_vector_type(8)))  float  v8f;
typedef __attribute__((ext_vector_type(4)))  int    int4v;

#define B_    32
#define T_    1024
#define H_    1024
#define LV_   128
#define NCOL  4352          // 4H + 2LV output columns of the fused projection
#define NWG   34            // 32 gate workgroups + 2 level workgroups
#define CHUNK 256           // K-chunk staged in LDS

#if __has_builtin(__builtin_amdgcn_global_load_async_to_lds_b128)
#define HAVE_ASYNC_LDS 1
#else
#define HAVE_ASYNC_LDS 0
#endif

// 16B global(bf16) -> LDS copy; async (ASYNCcnt) when the gfx1250 builtin exists.
// Signature (from compiler diagnostic): (int4 AS(1)*, int4 AS(3)*, imm offset, imm cpol).
__device__ __forceinline__ void copy16_g2l(const __bf16* gsrc, __bf16* ldst) {
#if HAVE_ASYNC_LDS
  __builtin_amdgcn_global_load_async_to_lds_b128(
      (__attribute__((address_space(1))) int4v*)gsrc,
      (__attribute__((address_space(3))) int4v*)ldst, 0, 0);
#else
  *(uint4*)ldst = *(const uint4*)gsrc;
#endif
}

__device__ __forceinline__ void wait_async_lds() {
#if HAVE_ASYNC_LDS
#if __has_builtin(__builtin_amdgcn_s_wait_asynccnt)
  __builtin_amdgcn_s_wait_asynccnt(0);
#else
  asm volatile("s_wait_asynccnt 0x0" ::: "memory");
#endif
#endif
}

// ---- A fragment: 16x32 bf16, per ISA 7.12.2: lane<16 -> K {0..7,16..23},
// lane>=16 -> K {8..15,24..31}; second run = +16 elems.
__device__ __forceinline__ v16bf make_afrag(const __bf16* p) {
  v8bf lo = *(const v8bf*)p;
  v8bf hi = *(const v8bf*)(p + 16);
  return __builtin_shufflevector(lo, hi, 0,1,2,3,4,5,6,7,8,9,10,11,12,13,14,15);
}

__device__ __forceinline__ float sigmoidf_(float x) { return 1.0f / (1.0f + __expf(-x)); }

// Generation-based grid barrier over NWG persistent blocks.
__device__ __forceinline__ void grid_barrier(unsigned* bar) {
  __syncthreads();
  if (threadIdx.x == 0) {
    __threadfence();
    unsigned* cnt = bar;
    unsigned* gen = bar + 1;
    unsigned g = __hip_atomic_load(gen, __ATOMIC_RELAXED, __HIP_MEMORY_SCOPE_AGENT);
    if (__hip_atomic_fetch_add(cnt, 1u, __ATOMIC_ACQ_REL, __HIP_MEMORY_SCOPE_AGENT) == NWG - 1u) {
      __hip_atomic_store(cnt, 0u, __ATOMIC_RELAXED, __HIP_MEMORY_SCOPE_AGENT);
      __hip_atomic_store(gen, g + 1u, __ATOMIC_RELEASE, __HIP_MEMORY_SCOPE_AGENT);
    } else {
      while (__hip_atomic_load(gen, __ATOMIC_ACQUIRE, __HIP_MEMORY_SCOPE_AGENT) == g)
        __builtin_amdgcn_s_sleep(2);
    }
    __threadfence();
  }
  __syncthreads();
}

// Pack [W_lstm ; W_lvl] -> bf16, row-major [NCOL][K].
__global__ void pack_weights_kernel(const float* __restrict__ Wl,
                                    const float* __restrict__ Wv,
                                    __bf16* __restrict__ Wp, int K) {
  size_t total = (size_t)NCOL * (size_t)K;
  for (size_t i = (size_t)blockIdx.x * blockDim.x + threadIdx.x; i < total;
       i += (size_t)gridDim.x * blockDim.x) {
    size_t row = i / (size_t)K;
    size_t k   = i - row * (size_t)K;
    float v = (row < 4096) ? Wl[row * (size_t)K + k]
                           : Wv[(row - 4096) * (size_t)K + k];
    Wp[i] = (__bf16)v;
  }
}

__global__ void cvt_f32_bf16_kernel(const float* __restrict__ src,
                                    __bf16* __restrict__ dst, size_t n) {
  for (size_t i = (size_t)blockIdx.x * blockDim.x + threadIdx.x; i < n;
       i += (size_t)gridDim.x * blockDim.x)
    dst[i] = (__bf16)src[i];
}

__global__ void init_state_kernel(__bf16* __restrict__ hbuf, unsigned* __restrict__ bar) {
  int i = blockIdx.x * blockDim.x + threadIdx.x;
  if (i < 2) bar[i] = 0u;
  for (int k = i; k < B_ * H_; k += gridDim.x * blockDim.x) hbuf[k] = (__bf16)0.f;
}

// Stage comb[:, ch*CHUNK : +CHUNK] into LDS buffer `buf` (async when available).
// All sources are bf16 in global memory: x part (pre-converted) or h buffer.
__device__ __forceinline__ void stage_chunk(const __bf16* __restrict__ xin,
                                            const __bf16* __restrict__ hbuf,
                                            __bf16 (*comb)[B_][CHUNK],
                                            int buf, int ch, int t, int Din, int tid) {
  const int kb = ch * CHUNK;
  for (int q = tid; q < B_ * (CHUNK / 8); q += 128) {   // 1024 x 16B segments
    const int row  = q >> 5;
    const int col0 = (q & 31) << 3;
    const int k0   = kb + col0;
    const __bf16* src = (k0 < Din)
        ? xin + ((size_t)row * T_ + t) * (size_t)Din + k0
        : hbuf + row * H_ + (k0 - Din);
    copy16_g2l(src, &comb[buf][row][col0]);
  }
}

// Persistent recurrent layer. 34 blocks x 128 threads (4 waves).
//  blocks 0..31: hidden-unit tile [b*32, b*32+32); wave w computes gate w (i/f/o/g).
//  block 32: cc_i_h (levels) -> master forget gate (backward cumsum of softmax)
//  block 33: cc_f_h (levels) -> master input gate  (forward  cumsum of softmax)
__global__ __launch_bounds__(128, 1) void onlstm_layer_kernel(
    const __bf16* __restrict__ xin, int Din,
    const __bf16* __restrict__ Wp,
    const float* __restrict__ bl, const float* __restrict__ bv,
    float* __restrict__ outf,          // fp32 output (final layer) or null
    __bf16* __restrict__ outbf,        // bf16 output (next layer's input) or null
    __bf16* __restrict__ hbuf,
    float* __restrict__ masters,       // [0..4095]=i_h, [4096..8191]=f_h   (32 x 128 each)
    unsigned* __restrict__ bar) {
  __shared__ __align__(32) __bf16 comb[2][B_][CHUNK];  // 32 KB double-buffered staging
  __shared__ float resbuf[4224];                       // gate WG: [4][32][33]; lvl WG: [32][132]

  const int tid  = threadIdx.x;
  const int lane = tid & 31;
  const int wv   = tid >> 5;
  const int lm   = lane & 15;
  const int lhi  = (lane >> 4) & 1;
  const int K    = Din + H_;
  const bool isGate = (blockIdx.x < 32);
  const int  wgbase = blockIdx.x * 32;               // hidden-unit tile base (gate WGs)
  const int  lw     = (int)blockIdx.x - 32;          // 0 -> cc_i_h, 1 -> cc_f_h
  const int  cbase  = isGate ? (wv * H_ + wgbase) : (4096 + lw * 128 + wv * 32);

  // Register-resident cell state: 32x32 tile, 8 elems per thread, fixed ownership.
  float creg[8];
#pragma unroll
  for (int e = 0; e < 8; ++e) creg[e] = 0.f;

  const float bias0 = isGate ? bl[cbase + lm]      : bv[cbase - 4096 + lm];
  const float bias1 = isGate ? bl[cbase + 16 + lm] : bv[cbase - 4096 + 16 + lm];

  for (int t = 0; t < T_; ++t) {
    v8f acc00 = {}, acc01 = {}, acc10 = {}, acc11 = {};
    const int nchunks = K / CHUNK;

    stage_chunk(xin, hbuf, comb, 0, 0, t, Din, tid);   // prologue: chunk 0 in flight

    for (int ch = 0; ch < nchunks; ++ch) {
      const int kb  = ch * CHUNK;
      const int buf = ch & 1;
      wait_async_lds();          // this wave's outstanding async LDS writes done
      __syncthreads();           // every wave's writes for chunk `ch` visible
      if (ch + 1 < nchunks)      // fire next chunk while we compute on this one
        stage_chunk(xin, hbuf, comb, (ch + 1) & 1, ch + 1, t, Din, tid);

      // GEMM over this K chunk: 2(M)x2(N) 16x16 tiles per wave, 8 k-steps of 32.
      const int klo = lhi * 8;
      const __bf16* wrow0 = Wp + (size_t)(cbase + lm) * (size_t)K + kb + lhi * 8;
      const __bf16* wrow1 = wrow0 + (size_t)16 * (size_t)K;
#pragma unroll
      for (int kc = 0; kc < CHUNK; kc += 32) {
        v16bf a0 = make_afrag(&comb[buf][lm][kc + klo]);
        v16bf a1 = make_afrag(&comb[buf][16 + lm][kc + klo]);
        v16bf b0 = make_afrag(wrow0 + kc);
        v16bf b1 = make_afrag(wrow1 + kc);
        acc00 = __builtin_amdgcn_wmma_f32_16x16x32_bf16(false, a0, false, b0, (short)0, acc00, false, false);
        acc01 = __builtin_amdgcn_wmma_f32_16x16x32_bf16(false, a0, false, b1, (short)0, acc01, false, false);
        acc10 = __builtin_amdgcn_wmma_f32_16x16x32_bf16(false, a1, false, b0, (short)0, acc10, false, false);
        acc11 = __builtin_amdgcn_wmma_f32_16x16x32_bf16(false, a1, false, b1, (short)0, acc11, false, false);
      }
    }

    // Spill projection tile (+bias) to LDS.
#pragma unroll
    for (int r = 0; r < 8; ++r) {
      const int m = lhi * 8 + r;  // D-matrix row per ISA layout: M = r + 8*(lane>=16)
      if (isGate) {
        resbuf[wv * 1056 + (m     ) * 33 + lm       ] = acc00[r] + bias0;
        resbuf[wv * 1056 + (m     ) * 33 + (16 + lm)] = acc01[r] + bias1;
        resbuf[wv * 1056 + (m + 16) * 33 + lm       ] = acc10[r] + bias0;
        resbuf[wv * 1056 + (m + 16) * 33 + (16 + lm)] = acc11[r] + bias1;
      } else {
        const int nb = wv * 32;
        resbuf[(m     ) * 132 + nb + lm       ] = acc00[r] + bias0;
        resbuf[(m     ) * 132 + nb + 16 + lm  ] = acc01[r] + bias1;
        resbuf[(m + 16) * 132 + nb + lm       ] = acc10[r] + bias0;
        resbuf[(m + 16) * 132 + nb + 16 + lm  ] = acc11[r] + bias1;
      }
    }

    if (!isGate) {
      __syncthreads();
      if (tid < B_) {  // one thread per batch row: softmax over 128 levels + cumsum
        float* row = &resbuf[tid * 132];
        float mx = -3.4e38f;
        for (int k2 = 0; k2 < 128; ++k2) mx = fmaxf(mx, row[k2]);
        float s = 0.f;
        for (int k2 = 0; k2 < 128; ++k2) { float e = __expf(row[k2] - mx); row[k2] = e; s += e; }
        const float inv = 1.f / s;
        float* mout = masters + (lw ? 0 : 4096) + tid * 128;
        if (lw == 1) {               // cc_f_h -> i_h : forward cumsum of softmax
          float c = 0.f;
          for (int k2 = 0; k2 < 128; ++k2) { c += row[k2] * inv; mout[k2] = c; }
        } else {                     // cc_i_h -> f_h : backward cumsum of softmax
          float c = 0.f;
          for (int k2 = 127; k2 >= 0; --k2) { c += row[k2] * inv; mout[k2] = c; }
        }
      }
    }

    grid_barrier(bar);  // masters published; all blocks done reading x_t / h_{t-1}

    if (isGate) {
#pragma unroll
      for (int e = 0; e < 8; ++e) {
        const int idx = tid + (e << 7);
        const int m = idx >> 5;      // batch row
        const int n = idx & 31;      // unit within tile
        const float gi = resbuf[0 * 1056 + m * 33 + n];
        const float gf = resbuf[1 * 1056 + m * 33 + n];
        const float go = resbuf[2 * 1056 + m * 33 + n];
        const float gg = resbuf[3 * 1056 + m * 33 + n];
        const float i = sigmoidf_(gi);
        const float f = sigmoidf_(gf);
        const float o = sigmoidf_(go);
        const float g = tanhf(gg);
        const int u  = wgbase + n;
        const int lv = u >> 3;       // NREP = 8
        const float ih = masters[m * 128 + lv];
        const float fh = masters[4096 + m * 128 + lv];
        const float w  = ih * fh;
        const float cn = w * (f * creg[e] + i * g) + (fh - w) * creg[e] + (ih - w) * g;
        const float hn = o * tanhf(cn);
        creg[e] = cn;
        const size_t oidx = ((size_t)m * T_ + t) * (size_t)H_ + u;
        if (outf)  outf[oidx]  = hn;
        if (outbf) outbf[oidx] = (__bf16)hn;
        hbuf[m * H_ + u] = (__bf16)hn;
      }
    }

    grid_barrier(bar);  // h_t visible everywhere before next step's staging
  }
}

extern "C" void kernel_launch(void* const* d_in, const int* in_sizes, int n_in,
                              void* d_out, int out_size, void* d_ws, size_t ws_size,
                              hipStream_t stream) {
  (void)in_sizes; (void)n_in; (void)out_size; (void)ws_size;
  const float* x   = (const float*)d_in[0];
  const float* Wl0 = (const float*)d_in[1];
  const float* bl0 = (const float*)d_in[2];
  const float* Wv0 = (const float*)d_in[3];
  const float* bv0 = (const float*)d_in[4];
  const float* Wl1 = (const float*)d_in[5];
  const float* bl1 = (const float*)d_in[6];
  const float* Wv1 = (const float*)d_in[7];
  const float* bv1 = (const float*)d_in[8];
  float* out = (float*)d_out;

  // Workspace carve (~132 MB, all offsets 256B-aligned).
  char* ws = (char*)d_ws;
  __bf16* Wp0 = (__bf16*)ws;   ws += (size_t)NCOL * 1536 * 2;
  __bf16* Wp1 = (__bf16*)ws;   ws += (size_t)NCOL * 2048 * 2;
  __bf16* hbuf = (__bf16*)ws;  ws += (size_t)B_ * H_ * 2;
  float* masters = (float*)ws; ws += (size_t)2 * B_ * LV_ * 4;
  unsigned* bar = (unsigned*)ws; ws += 256;
  __bf16* xbf0 = (__bf16*)ws;  ws += (size_t)B_ * T_ * 512 * 2;   // x in bf16
  __bf16* xbf1 = (__bf16*)ws;                                     // layer-0 output in bf16

  pack_weights_kernel<<<512, 256, 0, stream>>>(Wl0, Wv0, Wp0, 1536);
  pack_weights_kernel<<<512, 256, 0, stream>>>(Wl1, Wv1, Wp1, 2048);
  cvt_f32_bf16_kernel<<<512, 256, 0, stream>>>(x, xbf0, (size_t)B_ * T_ * 512);

  // Layer 0: xbf0 -> xbf1 (bf16 only; fp32 never touched on the staging path)
  init_state_kernel<<<64, 256, 0, stream>>>(hbuf, bar);
  onlstm_layer_kernel<<<NWG, 128, 0, stream>>>(xbf0, 512, Wp0, bl0, bv0,
                                               (float*)nullptr, xbf1, hbuf, masters, bar);

  // Layer 1: xbf1 -> d_out (fp32)
  init_state_kernel<<<64, 256, 0, stream>>>(hbuf, bar);
  onlstm_layer_kernel<<<NWG, 128, 0, stream>>>(xbf1, 1024, Wp1, bl1, bv1,
                                               out, (__bf16*)nullptr, hbuf, masters, bar);
}
